// MultiHeadSelfAttention_67053029425285
// MI455X (gfx1250) — matrix operand order repaired and checked by reading, not compiled, not run on hardware
//
#include <hip/hip_runtime.h>

// MI455X / gfx1250: wave32, WMMA 16x16x32 bf16 -> f32 accumulate.
#define BI 8
#define TI 1024
#define HN 8
#define HD 64

typedef __attribute__((ext_vector_type(16))) __bf16 v16bf;
typedef __attribute__((ext_vector_type(8)))  float  v8f;

union Frag { v16bf v; unsigned int u[8]; };

__device__ __forceinline__ unsigned short f2bf(float f) {
  unsigned int x = __builtin_bit_cast(unsigned int, f);
  x += 0x7fffu + ((x >> 16) & 1u);         // round-to-nearest-even
  return (unsigned short)(x >> 16);
}
__device__ __forceinline__ unsigned int packbf(float a, float b) {
  return (unsigned int)f2bf(a) | ((unsigned int)f2bf(b) << 16);
}
// K position of packed pair in 16-bit A-fragment (16x32), per ISA 7.12.2
__device__ __forceinline__ int kposA(int j, int hi) {
  return (j < 4 ? 2 * j : 16 + 2 * (j - 4)) + (hi ? 8 : 0);
}

// ---------------- GEMM: C[M,N] = A[M,512] @ W[512,N] + bias ----------------
// Block: 128 thr = 4 waves; tile 64x64; wave w -> rows 16w..16w+15, all 64 cols.
template <bool A_BF16, bool OUT_BF16>
__global__ __launch_bounds__(128) void gemm_bias_wmma(
    const void* __restrict__ Aptr, const float* __restrict__ W,
    const float* __restrict__ bias, void* __restrict__ Cptr, int N) {
  constexpr int K = 512;
  __shared__ alignas(16) unsigned short sWt[64 * 34];  // [n][k] transposed

  const int tid = threadIdx.x, wid = tid >> 5, lane = tid & 31;
  const int hi = lane >> 4, ln = lane & 15;
  const int row0 = blockIdx.x * 64 + wid * 16;
  const int col0 = blockIdx.y * 64;

  v8f acc[4];
#pragma unroll
  for (int t = 0; t < 4; ++t)
#pragma unroll
    for (int r = 0; r < 8; ++r) acc[t][r] = 0.0f;

  const float* Af = (const float*)Aptr;
  const unsigned short* Ab = (const unsigned short*)Aptr;
  const size_t arow = (size_t)(row0 + ln) * K;

  for (int kk = 0; kk < K; kk += 32) {
    __syncthreads();
    for (int i = tid; i < 32 * 64; i += 128) {
      int kr = i >> 6, nc = i & 63;
      sWt[nc * 34 + kr] = f2bf(W[(size_t)(kk + kr) * N + col0 + nc]);
    }
    __syncthreads();

    Frag a;
#pragma unroll
    for (int j = 0; j < 8; ++j) {
      int kp = kposA(j, hi) + kk;
      if constexpr (A_BF16)
        a.u[j] = *(const unsigned int*)(Ab + arow + kp);
      else
        a.u[j] = packbf(Af[arow + kp], Af[arow + kp + 1]);
    }
#pragma unroll
    for (int nt = 0; nt < 4; ++nt) {
      Frag b;
#pragma unroll
      for (int j = 0; j < 8; ++j)
        b.u[j] = *(const unsigned int*)(&sWt[(nt * 16 + ln) * 34 + kposA(j, hi)]);
      acc[nt] = __builtin_amdgcn_wmma_f32_16x16x32_bf16(
          false, a.v, false, b.v, (short)0, acc[nt], false, false);
    }
  }

#pragma unroll
  for (int nt = 0; nt < 4; ++nt)
#pragma unroll
    for (int r = 0; r < 8; ++r) {
      int row = row0 + r + hi * 8;
      int col = col0 + nt * 16 + ln;
      float vv = acc[nt][r] + bias[col];
      if constexpr (OUT_BF16)
        ((unsigned short*)Cptr)[(size_t)row * N + col] = f2bf(vv);
      else
        ((float*)Cptr)[(size_t)row * N + col] = vv;
    }
}

// ------------- Flash attention with relative-position bias -----------------
// Grid (T/64, H, B), 128 thr = 4 waves; wave w: 16 q rows. Key blocks of 32.
// pos_attn[n,m] = q[n]·rel_emb[clip(n-m,-512,512)+512]; for a 16x32 tile
// dist spans a 47-wide window -> compute qr = q @ rel_winT by WMMA, then
// per-element diagonal lookup qr[i, i-j+31] via LDS.
__global__ __launch_bounds__(128) void flash_attn_relpos(
    const unsigned short* __restrict__ qw, const unsigned short* __restrict__ kvw,
    const float* __restrict__ rel, unsigned short* __restrict__ ow) {
  __shared__ alignas(16) unsigned short sK[32 * 64];     // [key][d]
  __shared__ alignas(16) unsigned short sVt[64 * 36];    // [d][key]
  __shared__ alignas(16) unsigned short sR[112 * 64];    // rel window [u][d]
  __shared__ alignas(16) unsigned short sP[4 * 16 * 32]; // per-wave probs
  __shared__ alignas(16) float sQR[4 * 16 * 48];         // per-wave q·rel

  const int tid = threadIdx.x, wid = tid >> 5, lane = tid & 31;
  const int hi = lane >> 4, ln = lane & 15;
  const int qbase = blockIdx.x * 64;
  const int h = blockIdx.y, b = blockIdx.z;
  const int n0 = qbase + wid * 16;

  Frag a0, a1;  // q A-fragments (d = 0..31, 32..63)
  {
    const size_t qrow = ((size_t)(b * TI + n0 + ln)) * 512 + h * 64;
#pragma unroll
    for (int j = 0; j < 8; ++j) {
      int kp = kposA(j, hi);
      a0.u[j] = *(const unsigned int*)(qw + qrow + kp);
      a1.u[j] = *(const unsigned int*)(qw + qrow + kp + 32);
    }
  }

  v8f o[4];
#pragma unroll
  for (int t = 0; t < 4; ++t)
#pragma unroll
    for (int r = 0; r < 8; ++r) o[t][r] = 0.0f;
  float mst[8], lst[8];
#pragma unroll
  for (int r = 0; r < 8; ++r) { mst[r] = -1e30f; lst[r] = 0.0f; }

  for (int m0 = 0; m0 < TI; m0 += 32) {
    __syncthreads();
    for (int i = tid; i < 32 * 64; i += 128) {
      int kr = i >> 6, dc = i & 63;
      size_t base = ((size_t)(b * TI + m0 + kr)) * 1024 + h * 64 + dc;
      sK[kr * 64 + dc] = kvw[base];          // K half
      sVt[dc * 36 + kr] = kvw[base + 512];   // V half, transposed
    }
    const int wbase = qbase - m0 + 481;  // du = n-m+512, window starts at -31
    for (int i = tid; i < 112 * 64; i += 128) {
      int ur = i >> 6, dc = i & 63;
      int du = wbase + ur;
      du = du < 0 ? 0 : (du > 1024 ? 1024 : du);
      sR[ur * 64 + dc] = f2bf(rel[(size_t)du * 64 + dc]);
    }
    __syncthreads();

    // qr = q @ rel_winT (16 x 48), wave w uses window offset 16w
#pragma unroll
    for (int nt = 0; nt < 3; ++nt) {
      v8f qr = {0.f, 0.f, 0.f, 0.f, 0.f, 0.f, 0.f, 0.f};
      Frag b0, b1;
      int usel = wid * 16 + nt * 16 + ln;
#pragma unroll
      for (int j = 0; j < 8; ++j) {
        int d = kposA(j, hi);
        b0.u[j] = *(const unsigned int*)(&sR[usel * 64 + d]);
        b1.u[j] = *(const unsigned int*)(&sR[usel * 64 + d + 32]);
      }
      qr = __builtin_amdgcn_wmma_f32_16x16x32_bf16(false, a0.v, false, b0.v, (short)0, qr, false, false);
      qr = __builtin_amdgcn_wmma_f32_16x16x32_bf16(false, a1.v, false, b1.v, (short)0, qr, false, false);
#pragma unroll
      for (int r = 0; r < 8; ++r)
        sQR[(wid * 16 + r + hi * 8) * 48 + nt * 16 + ln] = qr[r];
    }

    // dots = q @ kT (16 x 32)
    v8f s[2];
#pragma unroll
    for (int nt2 = 0; nt2 < 2; ++nt2) {
#pragma unroll
      for (int r = 0; r < 8; ++r) s[nt2][r] = 0.0f;
      Frag b0, b1;
#pragma unroll
      for (int j = 0; j < 8; ++j) {
        int d = kposA(j, hi);
        b0.u[j] = *(const unsigned int*)(&sK[(nt2 * 16 + ln) * 64 + d]);
        b1.u[j] = *(const unsigned int*)(&sK[(nt2 * 16 + ln) * 64 + d + 32]);
      }
      s[nt2] = __builtin_amdgcn_wmma_f32_16x16x32_bf16(false, a0.v, false, b0.v, (short)0, s[nt2], false, false);
      s[nt2] = __builtin_amdgcn_wmma_f32_16x16x32_bf16(false, a1.v, false, b1.v, (short)0, s[nt2], false, false);
    }

    // online softmax over 16x32 block
#pragma unroll
    for (int r = 0; r < 8; ++r) {
      int mrow = r + hi * 8;
      const float* qrr = &sQR[(wid * 16 + mrow) * 48];
      float v0 = (s[0][r] + qrr[mrow - ln + 31]) * 0.125f;  // col j = ln
      float v1 = (s[1][r] + qrr[mrow - ln + 15]) * 0.125f;  // col j = ln+16
      float rm = fmaxf(v0, v1);
#pragma unroll
      for (int msk = 1; msk < 16; msk <<= 1) rm = fmaxf(rm, __shfl_xor(rm, msk, 16));
      float mn = fmaxf(mst[r], rm);
      float alpha = __expf(mst[r] - mn);
      float p0 = __expf(v0 - mn), p1 = __expf(v1 - mn);
      float rs = p0 + p1;
#pragma unroll
      for (int msk = 1; msk < 16; msk <<= 1) rs += __shfl_xor(rs, msk, 16);
      lst[r] = lst[r] * alpha + rs;
      mst[r] = mn;
#pragma unroll
      for (int nt = 0; nt < 4; ++nt) o[nt][r] *= alpha;
      sP[(wid * 16 + mrow) * 32 + ln] = f2bf(p0);
      sP[(wid * 16 + mrow) * 32 + ln + 16] = f2bf(p1);
    }

    // O += P @ V
    Frag pa;
#pragma unroll
    for (int j = 0; j < 8; ++j)
      pa.u[j] = *(const unsigned int*)(&sP[(wid * 16 + ln) * 32 + kposA(j, hi)]);
#pragma unroll
    for (int nt = 0; nt < 4; ++nt) {
      Frag bv;
#pragma unroll
      for (int j = 0; j < 8; ++j)
        bv.u[j] = *(const unsigned int*)(&sVt[(nt * 16 + ln) * 36 + kposA(j, hi)]);
      o[nt] = __builtin_amdgcn_wmma_f32_16x16x32_bf16(false, pa.v, false, bv.v, (short)0, o[nt], false, false);
    }
  }

#pragma unroll
  for (int r = 0; r < 8; ++r) {
    float inv = 1.0f / lst[r];
    int n = n0 + r + hi * 8;
    size_t base = ((size_t)(b * TI + n)) * 512 + h * 64;
#pragma unroll
    for (int nt = 0; nt < 4; ++nt)
      ow[base + nt * 16 + ln] = f2bf(o[nt][r] * inv);
  }
}

// ------------------------------ launch -------------------------------------
extern "C" void kernel_launch(void* const* d_in, const int* in_sizes, int n_in,
                              void* d_out, int out_size, void* d_ws, size_t ws_size,
                              hipStream_t stream) {
  (void)in_sizes; (void)n_in; (void)out_size; (void)ws_size;
  const float* x    = (const float*)d_in[0];
  const float* Wq   = (const float*)d_in[1];
  const float* bq   = (const float*)d_in[2];
  const float* Wkv  = (const float*)d_in[3];
  const float* bkv  = (const float*)d_in[4];
  const float* Wout = (const float*)d_in[5];
  const float* bout = (const float*)d_in[6];
  const float* rel  = (const float*)d_in[7];
  float* out = (float*)d_out;

  unsigned short* qws  = (unsigned short*)d_ws;                // 8192x512 bf16
  unsigned short* kvws = qws + (size_t)8192 * 512;             // 8192x1024 bf16
  unsigned short* aws  = kvws + (size_t)8192 * 1024;           // 8192x512 bf16

  dim3 blk(128);
  gemm_bias_wmma<false, true><<<dim3(128, 8),  blk, 0, stream>>>(x,   Wq,   bq,   qws,  512);
  gemm_bias_wmma<false, true><<<dim3(128, 16), blk, 0, stream>>>(x,   Wkv,  bkv,  kvws, 1024);
  flash_attn_relpos<<<dim3(TI / 64, HN, BI), blk, 0, stream>>>(qws, kvws, rel, aws);
  gemm_bias_wmma<true, false><<<dim3(128, 8),  blk, 0, stream>>>(aws, Wout, bout, out,  512);
}
